// Router_15058155340099
// MI455X (gfx1250) — matrix operand order repaired and checked by reading, not compiled, not run on hardware
//
#include <hip/hip_runtime.h>
#include <stdint.h>

// ---------------- types for WMMA ----------------
typedef __attribute__((ext_vector_type(16))) __bf16 v16bf;
typedef __attribute__((ext_vector_type(8)))  float  v8f;

#define T_TOKENS 8192
#define D_DIM    4096
#define E_DIM    64
#define TOPK     8

#define KC            256          // K-chunk staged in LDS per iteration
#define LDS_PAIR_ROWS (KC / 2)     // 128 pair-rows (each holds k even/odd bf16 pair)
#define LDS_PITCH     65           // padded row pitch in dwords (bank-conflict avoidance)

union BFragU { v16bf bf; uint32_t u[8]; };
union PairU  { __bf16 h[2]; uint32_t u; };

// =====================================================================
// Kernel 1: router logits GEMM via v_wmma_f32_16x16x32_bf16
//   grid = 128 blocks * 128 threads (4 waves). Wave w computes tokens
//   [blk*64 + w*16, +16) x all 64 experts. Kernel matrix staged in LDS.
// =====================================================================
__global__ __launch_bounds__(128) void router_gemm_wmma(
    const float* __restrict__ x,      // (8192, 4096) row-major f32
    const float* __restrict__ wDE,    // (4096, 64)   row-major f32
    float* __restrict__ logits)       // (8192, 64)   f32
{
  __shared__ uint32_t sB[LDS_PAIR_ROWS * LDS_PITCH];   // 33,280 bytes

  const int tid   = threadIdx.x;
  const int lane  = tid & 31;
  const int wave  = tid >> 5;
  const int tokBase = blockIdx.x * 64 + wave * 16;

  const int m     = lane & 15;       // token row within tile
  const int khalf = lane >> 4;       // 0 or 1 (K half, per ISA A/B layouts)
  const float* __restrict__ xrow = x + (size_t)(tokBase + m) * D_DIM;

  v8f acc0 = {}, acc1 = {}, acc2 = {}, acc3 = {};

  for (int kb = 0; kb < D_DIM; kb += KC) {
    __syncthreads();
    // ---- producer: stage wDE[kb .. kb+KC) x 64 into LDS as bf16 pairs ----
    // pair-row kp holds (k = kb+2*kp, k+1); col e. Thread t does i = t + 128*j.
    for (int j = 0; j < (LDS_PAIR_ROWS * E_DIM) / 128; ++j) {
      int i  = tid + 128 * j;
      int kp = i >> 6;
      int e  = i & 63;
      size_t gk = (size_t)(kb + kp * 2) * E_DIM + e;
      float f0 = wDE[gk];
      float f1 = wDE[gk + E_DIM];
      PairU p; p.h[0] = (__bf16)f0; p.h[1] = (__bf16)f1;
      sB[kp * LDS_PITCH + e] = p.u;
    }
    __syncthreads();

    // ---- consumer: 8 wmma K-steps of 32 over this chunk ----
    for (int kk = 0; kk < KC; kk += 32) {
      // A fragment (16x32 bf16): lane<16 -> K {k0..k0+7, k0+16..k0+23},
      // lanes 16-31 shifted by +8 via khalf.
      const int k0 = kb + kk + khalf * 8;
      float4 f0 = *(const float4*)(xrow + k0);
      float4 f1 = *(const float4*)(xrow + k0 + 4);
      float4 f2 = *(const float4*)(xrow + k0 + 16);
      float4 f3 = *(const float4*)(xrow + k0 + 20);
      v16bf a;
      a[0]=(__bf16)f0.x;  a[1]=(__bf16)f0.y;  a[2]=(__bf16)f0.z;  a[3]=(__bf16)f0.w;
      a[4]=(__bf16)f1.x;  a[5]=(__bf16)f1.y;  a[6]=(__bf16)f1.z;  a[7]=(__bf16)f1.w;
      a[8]=(__bf16)f2.x;  a[9]=(__bf16)f2.y;  a[10]=(__bf16)f2.z; a[11]=(__bf16)f2.w;
      a[12]=(__bf16)f3.x; a[13]=(__bf16)f3.y; a[14]=(__bf16)f3.z; a[15]=(__bf16)f3.w;

      // B fragments (32x16 bf16 each): lane holds column N=lane%16;
      // VGPR v<4 holds K pair (2v,2v+1), v>=4 holds K pair (2v+8,2v+9) (+8 for upper lane half).
      const int kpBase = (kk >> 1) + khalf * 4;
      const int eLane  = lane & 15;
      BFragU b0, b1, b2, b3;
      #pragma unroll
      for (int v = 0; v < 8; ++v) {
        int row = (kpBase + v + (v >= 4 ? 4 : 0)) * LDS_PITCH + eLane;
        b0.u[v] = sB[row +  0];
        b1.u[v] = sB[row + 16];
        b2.u[v] = sB[row + 32];
        b3.u[v] = sB[row + 48];
      }
      acc0 = __builtin_amdgcn_wmma_f32_16x16x32_bf16(false, a, false, b0.bf, (short)0, acc0, false, false);
      acc1 = __builtin_amdgcn_wmma_f32_16x16x32_bf16(false, a, false, b1.bf, (short)0, acc1, false, false);
      acc2 = __builtin_amdgcn_wmma_f32_16x16x32_bf16(false, a, false, b2.bf, (short)0, acc2, false, false);
      acc3 = __builtin_amdgcn_wmma_f32_16x16x32_bf16(false, a, false, b3.bf, (short)0, acc3, false, false);
    }
  }

  // C/D layout: VGPR j, lanes 0-15 -> M=j, lanes 16-31 -> M=j+8; N = lane%16.
  const int tokOff = tokBase + khalf * 8;
  const int eLane  = lane & 15;
  #pragma unroll
  for (int j = 0; j < 8; ++j) {
    size_t base = (size_t)(tokOff + j) * E_DIM + eLane;
    logits[base +  0] = acc0[j];
    logits[base + 16] = acc1[j];
    logits[base + 32] = acc2[j];
    logits[base + 48] = acc3[j];
  }
}

// =====================================================================
// Kernel 2: per-token top-8 + softmax. One token per thread.
// Matches lax.top_k ordering: descending values, lowest index on ties.
// =====================================================================
__global__ __launch_bounds__(256) void router_topk_softmax(
    const float* __restrict__ logits,   // (8192, 64)
    float* __restrict__ wOut,           // (8192, 8)
    int*   __restrict__ iOut)           // (8192, 8)
{
  const int token = blockIdx.x * 256 + threadIdx.x;
  if (token >= T_TOKENS) return;

  float v[E_DIM];
  const float4* row = (const float4*)(logits + (size_t)token * E_DIM);
  #pragma unroll
  for (int i = 0; i < E_DIM / 4; ++i) {
    float4 q = row[i];
    v[4*i+0] = q.x; v[4*i+1] = q.y; v[4*i+2] = q.z; v[4*i+3] = q.w;
  }

  float wsel[TOPK];
  int   isel[TOPK];
  #pragma unroll
  for (int s = 0; s < TOPK; ++s) {
    float best = -INFINITY;
    int   bi   = 0;
    #pragma unroll
    for (int e = 0; e < E_DIM; ++e) {
      bool c = v[e] > best;           // strict > + ascending scan => lowest-index tie-break
      best = c ? v[e] : best;
      bi   = c ? e    : bi;
    }
    wsel[s] = best;
    isel[s] = bi;
    #pragma unroll
    for (int e = 0; e < E_DIM; ++e)
      v[e] = (e == bi) ? -INFINITY : v[e];
  }

  // softmax over the 8 selected (values are descending => max = wsel[0])
  const float mx = wsel[0];
  float ex[TOPK];
  float sum = 0.0f;
  #pragma unroll
  for (int s = 0; s < TOPK; ++s) { ex[s] = __expf(wsel[s] - mx); sum += ex[s]; }
  const float inv = 1.0f / sum;
  #pragma unroll
  for (int s = 0; s < TOPK; ++s) {
    wOut[(size_t)token * TOPK + s] = ex[s] * inv;
    iOut[(size_t)token * TOPK + s] = isel[s];
  }
}

// =====================================================================
extern "C" void kernel_launch(void* const* d_in, const int* in_sizes, int n_in,
                              void* d_out, int out_size, void* d_ws, size_t ws_size,
                              hipStream_t stream) {
  const float* x   = (const float*)d_in[0];   // (8192, 4096) f32
  const float* wDE = (const float*)d_in[1];   // (4096, 64)   f32

  float* logits = (float*)d_ws;               // 8192*64*4 = 2 MB scratch

  float* wOut = (float*)d_out;                             // first output: weights
  int*   iOut = (int*)((float*)d_out + T_TOKENS * TOPK);   // second output: indices

  router_gemm_wmma<<<dim3(T_TOKENS / 64), dim3(128), 0, stream>>>(x, wDE, logits);
  router_topk_softmax<<<dim3(T_TOKENS / 256), dim3(256), 0, stream>>>(logits, wOut, iOut);
}
